// GQAAttention_71167608094828
// MI455X (gfx1250) — matrix operand order repaired and checked by reading, not compiled, not run on hardware
//
#include <hip/hip_runtime.h>
#include <hip/hip_bf16.h>

// ---------------------------------------------------------------------------
// Types for CDNA5 WMMA (wave32). bf16 16x16x32, fp32 accumulate.
// ---------------------------------------------------------------------------
typedef __bf16 bf16;
typedef __attribute__((ext_vector_type(16))) __bf16 v16bf;
typedef __attribute__((ext_vector_type(8)))  __bf16 bf16x8;
typedef __attribute__((ext_vector_type(4)))  __bf16 bf16x4;
typedef __attribute__((ext_vector_type(8)))  float  v8f;

union AFrag {
    v16bf  v;
    bf16x8 h[2];
};

static constexpr int S_LEN  = 2048;
static constexpr int HIDDEN = 2048;
static constexpr int NHEADS = 32;
static constexpr int NKV    = 8;
static constexpr int HDIM   = 64;
static constexpr int KVW    = NKV * HDIM;     // 512
static constexpr int MROWS  = 2 * S_LEN;      // 4096 (B*S)

// ---------------------------------------------------------------------------
// Cross-lane helpers: butterfly reductions over the 16-lane half-rows that
// hold one matrix row of a WMMA C tile. Prefer v_permlane16 (VALU, co-executes
// with WMMA on the XDL pipe) over ds_bpermute (burns LDS bandwidth).
// ---------------------------------------------------------------------------
#if __has_builtin(__builtin_amdgcn_permlane16)
#define USE_PERMLANE16 1
#endif

#if defined(USE_PERMLANE16)
__device__ __forceinline__ float permxor16(float v, unsigned lo, unsigned hi)
{
    int i = __float_as_int(v);
    int r = __builtin_amdgcn_permlane16(i, i, (int)lo, (int)hi, false, false);
    return __int_as_float(r);
}
__device__ __forceinline__ float rowmax16(float v)
{
    v = fmaxf(v, permxor16(v, 0x67452301u, 0xEFCDAB89u));  // xor 1
    v = fmaxf(v, permxor16(v, 0x54761032u, 0xDCFE98BAu));  // xor 2
    v = fmaxf(v, permxor16(v, 0x32107654u, 0xBA98FEDCu));  // xor 4
    v = fmaxf(v, permxor16(v, 0xFEDCBA98u, 0x76543210u));  // xor 8
    return v;
}
__device__ __forceinline__ float rowsum16(float v)
{
    v += permxor16(v, 0x67452301u, 0xEFCDAB89u);
    v += permxor16(v, 0x54761032u, 0xDCFE98BAu);
    v += permxor16(v, 0x32107654u, 0xBA98FEDCu);
    v += permxor16(v, 0xFEDCBA98u, 0x76543210u);
    return v;
}
#else
__device__ __forceinline__ float rowmax16(float v)
{
#pragma unroll
    for (int m = 1; m <= 8; m <<= 1) v = fmaxf(v, __shfl_xor(v, m, 32));
    return v;
}
__device__ __forceinline__ float rowsum16(float v)
{
#pragma unroll
    for (int m = 1; m <= 8; m <<= 1) v += __shfl_xor(v, m, 32);
    return v;
}
#endif

// ---------------------------------------------------------------------------
// Async global->LDS staging (CDNA5 GLOBAL_LOAD_ASYNC_TO_LDS_*, ASYNCcnt).
// Builtin signature (from hipcc diagnostic): arg0 = int4* in the device-global
// address space (non-const), arg1 = LDS-side pointer, then imm offset, cpol.
// ---------------------------------------------------------------------------
#if __has_builtin(__builtin_amdgcn_global_load_async_to_lds_b128)
#define USE_ASYNC_LDS 1
typedef int v4i __attribute__((ext_vector_type(4)));
typedef __attribute__((address_space(1))) v4i* gas4;
typedef __attribute__((address_space(3))) v4i* las4;
__device__ __forceinline__ void async_copy16(const bf16* g, bf16* l)
{
    __builtin_amdgcn_global_load_async_to_lds_b128(
        (gas4)(v4i*)g, (las4)(v4i*)l, 0, 0);
}
__device__ __forceinline__ void async_wait0()
{
#if __has_builtin(__builtin_amdgcn_s_wait_asynccnt)
    __builtin_amdgcn_s_wait_asynccnt(0);
#else
    asm volatile("s_wait_asynccnt 0" ::: "memory");
#endif
}
#endif

// ---------------------------------------------------------------------------
// fp32 -> bf16 elementwise
// ---------------------------------------------------------------------------
__global__ __launch_bounds__(256)
void f32_to_bf16_k(const float* __restrict__ in, bf16* __restrict__ out, int n)
{
    int i = (int)(blockIdx.x * 256u + threadIdx.x) * 4;
    if (i + 3 < n) {
        float4 f = *(const float4*)(in + i);
        bf16x4 o;
        o[0] = (bf16)f.x; o[1] = (bf16)f.y; o[2] = (bf16)f.z; o[3] = (bf16)f.w;
        *(bf16x4*)(out + i) = o;
    }
}

// ---------------------------------------------------------------------------
// fp32 [K][N] -> bf16 [N][K] tiled transpose+convert (weights).
// ---------------------------------------------------------------------------
__global__ __launch_bounds__(256)
void transpose_bf16_k(const float* __restrict__ W, bf16* __restrict__ WT, int K, int N)
{
    __shared__ float tile[32][33];
    const int n0 = blockIdx.x * 32;
    const int k0 = blockIdx.y * 32;
    const int tx = threadIdx.x & 31;
    const int ty = threadIdx.x >> 5;      // 0..7
#pragma unroll
    for (int i = 0; i < 32; i += 8)
        tile[ty + i][tx] = W[(size_t)(k0 + ty + i) * N + (n0 + tx)];
    __syncthreads();
#pragma unroll
    for (int i = 0; i < 32; i += 8)
        WT[(size_t)(n0 + ty + i) * K + (k0 + tx)] = (bf16)tile[tx][ty + i];
}

// ---------------------------------------------------------------------------
// WMMA GEMM: C[M,N] = A[M,K](bf16,row) * BT[N,K](bf16,row)^T, fp32 accumulate.
// Block = 256 threads = 8 waves arranged 4(M) x 2(N); wave tile 32x64;
// block tile 128x128. 8 WMMAs per K-step per wave for 192B/lane of loads.
// ---------------------------------------------------------------------------
template <bool F32OUT>
__global__ __launch_bounds__(256)
void gemm_wmma_bf16(const bf16* __restrict__ A, const bf16* __restrict__ BT,
                    void* __restrict__ C, int M, int N, int K)
{
    const int lane = threadIdx.x & 31;
    const int wave = threadIdx.x >> 5;
    const int hi   = lane >> 4;       // 0/1: 16-lane half
    const int l15  = lane & 15;
    const int wm   = wave & 3;        // 0..3
    const int wn   = wave >> 2;       // 0..1
    const int mBase = blockIdx.x * 128 + wm * 32;
    const int nBase = blockIdx.y * 128 + wn * 64;

    v8f acc[2][4];
#pragma unroll
    for (int mi = 0; mi < 2; ++mi)
#pragma unroll
        for (int ni = 0; ni < 4; ++ni) acc[mi][ni] = (v8f){};

    const bf16* aR[2];
    const bf16* bR[4];
#pragma unroll
    for (int mi = 0; mi < 2; ++mi)
        aR[mi] = A + (size_t)(mBase + mi * 16 + l15) * K + 16 * hi;
#pragma unroll
    for (int ni = 0; ni < 4; ++ni)
        bR[ni] = BT + (size_t)(nBase + ni * 16 + l15) * K + 16 * hi;

#pragma unroll 2
    for (int kb = 0; kb < K; kb += 32) {
        AFrag a[2];
        v16bf b[4];
#pragma unroll
        for (int mi = 0; mi < 2; ++mi) {
            a[mi].v = *(const v16bf*)(aR[mi] + kb);
        }
#pragma unroll
        for (int ni = 0; ni < 4; ++ni)
            b[ni] = *(const v16bf*)(bR[ni] + kb);

        __builtin_prefetch((const void*)(aR[0] + kb + 32), 0, 0);
        __builtin_prefetch((const void*)(bR[0] + kb + 32), 0, 0);
        __builtin_prefetch((const void*)(bR[2] + kb + 32), 0, 0);

#pragma unroll
        for (int mi = 0; mi < 2; ++mi)
#pragma unroll
            for (int ni = 0; ni < 4; ++ni)
                acc[mi][ni] = __builtin_amdgcn_wmma_f32_16x16x32_bf16(
                    false, a[mi].v, false, b[ni], (short)0, acc[mi][ni],
                    false, false);
    }

    // C layout: VGPR r holds M = r + 8*hi (per half), N = l15.
#pragma unroll
    for (int mi = 0; mi < 2; ++mi)
#pragma unroll
        for (int ni = 0; ni < 4; ++ni) {
            const int col = nBase + ni * 16 + l15;
#pragma unroll
            for (int r = 0; r < 8; ++r) {
                const int row = mBase + mi * 16 + r + 8 * hi;
                const float v = acc[mi][ni][r];
                if (F32OUT) ((float*)C)[(size_t)row * N + col] = v;
                else        ((bf16*)C)[(size_t)row * N + col] = (bf16)v;
            }
        }
}

// ---------------------------------------------------------------------------
// Flash-attention (GQA): one block = 64 query rows of one (batch, head).
// 4 waves x 16 query rows. Streams 32-key blocks through LDS.
// ---------------------------------------------------------------------------
static constexpr int KPAD = 80;   // K tile row stride (bf16) -> 160B rows
static constexpr int VPAD = 48;   // Vt tile row stride      ->  96B rows
static constexpr int PPAD = 48;   // P tile row stride

__global__ __launch_bounds__(128)
void attn_wmma_k(const bf16* __restrict__ Qb, const bf16* __restrict__ Kb,
                 const bf16* __restrict__ Vb, bf16* __restrict__ Oa)
{
    const int qt   = blockIdx.x;          // query tile (64 rows)
    const int h    = blockIdx.y;          // query head
    const int b    = blockIdx.z;          // batch
    const int kvh  = h >> 2;              // 4 query heads per kv head
    const int lane = threadIdx.x & 31;
    const int wave = threadIdx.x >> 5;
    const int hi   = lane >> 4;
    const int l15  = lane & 15;
    const int qRow0 = qt * 64 + wave * 16;

    __shared__ bf16 Klds[32][KPAD];           // keys x d   (natural)
    __shared__ bf16 Vtlds[HDIM][VPAD];        // d x keys   (transposed)
    __shared__ bf16 Plds[4][16][PPAD];        // per-wave P staging

    // Q fragments for the two d-chunks (d = 64 = 2 x 32)
    AFrag aq[2];
    {
        const bf16* qp = Qb + (size_t)(b * S_LEN + qRow0 + l15) * (NHEADS * HDIM)
                            + h * HDIM + 16 * hi;
#pragma unroll
        for (int c = 0; c < 2; ++c) {
            aq[c].v = *(const v16bf*)(qp + c * 32);
        }
    }

    float mrun[8], lrun[8];
#pragma unroll
    for (int r = 0; r < 8; ++r) { mrun[r] = -3.0e30f; lrun[r] = 0.0f; }
    v8f o[4];
#pragma unroll
    for (int dt = 0; dt < 4; ++dt) o[dt] = (v8f){};

    const float scale = 0.125f;   // 1/sqrt(64)

    const int row = threadIdx.x >> 2;       // 0..31 (coop-load row)
    const int seg = threadIdx.x & 3;        // 16-wide column segment

    for (int kb0 = 0; kb0 < S_LEN; kb0 += 32) {
        // ---- cooperative staging of 32x64 K (async if available) and V ----
        {
            const size_t src = (size_t)(b * S_LEN + kb0 + row) * KVW
                             + kvh * HDIM + seg * 16;
#if defined(USE_ASYNC_LDS)
            async_copy16(Kb + src,     &Klds[row][seg * 16]);
            async_copy16(Kb + src + 8, &Klds[row][seg * 16 + 8]);
#else
            bf16x8 k0 = *(const bf16x8*)(Kb + src);
            bf16x8 k1 = *(const bf16x8*)(Kb + src + 8);
            *(bf16x8*)&Klds[row][seg * 16]     = k0;
            *(bf16x8*)&Klds[row][seg * 16 + 8] = k1;
#endif
            bf16x8 v0 = *(const bf16x8*)(Vb + src);
            bf16x8 v1 = *(const bf16x8*)(Vb + src + 8);
#pragma unroll
            for (int j = 0; j < 8; ++j) {
                Vtlds[seg * 16 + j][row]     = v0[j];
                Vtlds[seg * 16 + 8 + j][row] = v1[j];
            }
#if defined(USE_ASYNC_LDS)
            async_wait0();
#endif
        }
        __syncthreads();

        // ---- scores: S(16x32) = Q(16x64) @ K^T ----
        v8f sc[2];
        sc[0] = (v8f){}; sc[1] = (v8f){};
#pragma unroll
        for (int nt = 0; nt < 2; ++nt)
#pragma unroll
            for (int c = 0; c < 2; ++c) {
                v16bf bk = *(const v16bf*)&Klds[nt * 16 + l15][c * 32 + 16 * hi];
                sc[nt] = __builtin_amdgcn_wmma_f32_16x16x32_bf16(
                    false, aq[c].v, false, bk, (short)0, sc[nt], false, false);
            }

        // ---- online softmax (row = r + 8*hi; reduce across 16-lane half) ----
        float corr[8];
#pragma unroll
        for (int r = 0; r < 8; ++r) {
            float s0 = sc[0][r] * scale;
            float s1 = sc[1][r] * scale;
            float mx = rowmax16(fmaxf(s0, s1));
            const float mnew = fmaxf(mrun[r], mx);
            corr[r] = __expf(mrun[r] - mnew);
            const float p0 = __expf(s0 - mnew);
            const float p1 = __expf(s1 - mnew);
            const float rs = rowsum16(p0 + p1);
            lrun[r] = lrun[r] * corr[r] + rs;
            mrun[r] = mnew;
            sc[0][r] = p0;
            sc[1][r] = p1;
        }
#pragma unroll
        for (int dt = 0; dt < 4; ++dt)
#pragma unroll
            for (int r = 0; r < 8; ++r) o[dt][r] *= corr[r];

        // ---- P: C-layout -> bf16 -> LDS -> A-layout fragment ----
#pragma unroll
        for (int nt = 0; nt < 2; ++nt)
#pragma unroll
            for (int r = 0; r < 8; ++r)
                Plds[wave][r + 8 * hi][nt * 16 + l15] = (bf16)sc[nt][r];
        __syncthreads();

        AFrag ap;
        {
            const bf16* pp = &Plds[wave][l15][16 * hi];
            ap.v = *(const v16bf*)pp;
        }

        // ---- O += P(16x32) @ V(32x64) ----
#pragma unroll
        for (int dt = 0; dt < 4; ++dt) {
            v16bf bv = *(const v16bf*)&Vtlds[dt * 16 + l15][16 * hi];
            o[dt] = __builtin_amdgcn_wmma_f32_16x16x32_bf16(
                false, ap.v, false, bv, (short)0, o[dt], false, false);
        }
        __syncthreads();   // protect K/V/P tiles before next block
    }

    // ---- epilogue: normalize and store bf16 attention output ----
#pragma unroll
    for (int dt = 0; dt < 4; ++dt)
#pragma unroll
        for (int r = 0; r < 8; ++r) {
            const float inv = 1.0f / lrun[r];
            const size_t orow = (size_t)(b * S_LEN + qRow0 + r + 8 * hi);
            Oa[orow * (NHEADS * HDIM) + h * HDIM + dt * 16 + l15] =
                (bf16)(o[dt][r] * inv);
        }
}

// ---------------------------------------------------------------------------
// Host-side launch
// ---------------------------------------------------------------------------
extern "C" void kernel_launch(void* const* d_in, const int* in_sizes, int n_in,
                              void* d_out, int out_size, void* d_ws, size_t ws_size,
                              hipStream_t stream)
{
    const float* x  = (const float*)d_in[0];
    const float* wq = (const float*)d_in[1];
    const float* wk = (const float*)d_in[2];
    const float* wv = (const float*)d_in[3];
    const float* wo = (const float*)d_in[4];
    float* out = (float*)d_out;

    char* ws = (char*)d_ws;
    size_t off = 0;
    auto alloc = [&](size_t bytes) -> void* {
        void* p = ws + off;
        off = (off + bytes + 255) & ~(size_t)255;
        return p;
    };

    bf16* xbf  = (bf16*)alloc((size_t)MROWS * HIDDEN * 2);
    bf16* wqT  = (bf16*)alloc((size_t)HIDDEN * HIDDEN * 2);
    bf16* wkT  = (bf16*)alloc((size_t)KVW * HIDDEN * 2);
    bf16* wvT  = (bf16*)alloc((size_t)KVW * HIDDEN * 2);
    bf16* woT  = (bf16*)alloc((size_t)HIDDEN * HIDDEN * 2);
    bf16* Qb   = (bf16*)alloc((size_t)MROWS * HIDDEN * 2);
    bf16* Kb   = (bf16*)alloc((size_t)MROWS * KVW * 2);
    bf16* Vb   = (bf16*)alloc((size_t)MROWS * KVW * 2);
    bf16* attn = (bf16*)alloc((size_t)MROWS * HIDDEN * 2);

    // 1) convert x to bf16
    {
        const int n = MROWS * HIDDEN;
        f32_to_bf16_k<<<dim3(n / (256 * 4)), dim3(256), 0, stream>>>(x, xbf, n);
    }
    // 2) convert + transpose weights: W[K][N] -> WT[N][K]
    transpose_bf16_k<<<dim3(HIDDEN / 32, HIDDEN / 32), dim3(256), 0, stream>>>(wq, wqT, HIDDEN, HIDDEN);
    transpose_bf16_k<<<dim3(KVW / 32,    HIDDEN / 32), dim3(256), 0, stream>>>(wk, wkT, HIDDEN, KVW);
    transpose_bf16_k<<<dim3(KVW / 32,    HIDDEN / 32), dim3(256), 0, stream>>>(wv, wvT, HIDDEN, KVW);
    transpose_bf16_k<<<dim3(HIDDEN / 32, HIDDEN / 32), dim3(256), 0, stream>>>(wo, woT, HIDDEN, HIDDEN);

    // 3) Q/K/V projections (bf16 out)
    gemm_wmma_bf16<false><<<dim3(MROWS / 128, HIDDEN / 128), dim3(256), 0, stream>>>(
        xbf, wqT, (void*)Qb, MROWS, HIDDEN, HIDDEN);
    gemm_wmma_bf16<false><<<dim3(MROWS / 128, KVW / 128), dim3(256), 0, stream>>>(
        xbf, wkT, (void*)Kb, MROWS, KVW, HIDDEN);
    gemm_wmma_bf16<false><<<dim3(MROWS / 128, KVW / 128), dim3(256), 0, stream>>>(
        xbf, wvT, (void*)Vb, MROWS, KVW, HIDDEN);

    // 4) GQA flash attention
    attn_wmma_k<<<dim3(S_LEN / 64, NHEADS, 2), dim3(128), 0, stream>>>(Qb, Kb, Vb, attn);

    // 5) output projection (fp32 out -> d_out)
    gemm_wmma_bf16<true><<<dim3(MROWS / 128, HIDDEN / 128), dim3(256), 0, stream>>>(
        attn, woT, (void*)out, MROWS, HIDDEN, HIDDEN);

    (void)in_sizes; (void)n_in; (void)out_size; (void)ws_size;
}
